// ComplexMoE_24902220382474
// MI455X (gfx1250) — matrix operand order repaired and checked by reading, not compiled, not run on hardware
//
#include <hip/hip_runtime.h>
#include <hip/hip_bf16.h>

#define DIM   1024
#define NEXP  8
#define NTOK  8192

typedef __attribute__((ext_vector_type(16))) __bf16 v16bf;
typedef __attribute__((ext_vector_type(8)))  float  v8f;
typedef unsigned int  u32;
typedef unsigned short u16;

// ---------- fp32 -> bf16 (RNE) packing ----------
__device__ __forceinline__ u32 pack_bf2(float lo, float hi) {
    u32 a = __builtin_bit_cast(u32, lo);
    u32 b = __builtin_bit_cast(u32, hi);
    a += 0x7FFFu + ((a >> 16) & 1u);
    b += 0x7FFFu + ((b >> 16) & 1u);
    return (a >> 16) | (b & 0xFFFF0000u);
}

// ---------- WMMA operand gathers from LDS (row-major [16][K] tiles) ----------
// A operand (16-bit A 16x32, ISA 7.12.2): lane L -> row M=L&15,
//   K = {kb..kb+7} in VGPR0..3 and {kb+16..kb+23} in VGPR4..7, kb = (L<16)?0:8.
__device__ __forceinline__ v16bf ldfragA(const u16* p) {
    union { u32 u[8]; v16bf v; } f;
    uint4 lo = *(const uint4*)p;          // K = kb .. kb+7
    uint4 hi = *(const uint4*)(p + 16);   // K = kb+16 .. kb+23
    f.u[0]=lo.x; f.u[1]=lo.y; f.u[2]=lo.z; f.u[3]=lo.w;
    f.u[4]=hi.x; f.u[5]=hi.y; f.u[6]=hi.z; f.u[7]=hi.w;
    return f.v;
}
// B operand (16-bit B 32x16): lane L -> column N=L&15,
//   K = {kb..kb+7} in VGPR0..3 and {kb+16..kb+23} in VGPR4..7, kb = (L<16)?0:8.
__device__ __forceinline__ v16bf ldfragB(const u16* p) {
    union { u32 u[8]; v16bf v; } f;
    uint4 lo = *(const uint4*)p;          // K = kb .. kb+7
    uint4 hi = *(const uint4*)(p + 16);   // K = kb+16 .. kb+23
    f.u[0]=lo.x; f.u[1]=lo.y; f.u[2]=lo.z; f.u[3]=lo.w;
    f.u[4]=hi.x; f.u[5]=hi.y; f.u[6]=hi.z; f.u[7]=hi.w;
    return f.v;
}

// ================= gate kernel: softmax(z_flat @ Wg^T + bg) =================
__global__ __launch_bounds__(256) void gate_kernel(
    const float* __restrict__ zr, const float* __restrict__ zi,
    const float* __restrict__ Wg, const float* __restrict__ bg,
    float* __restrict__ gates)
{
    const int lane = threadIdx.x & 31;
    const int wave = threadIdx.x >> 5;
    const int token = blockIdx.x * 8 + wave;   // one wave32 per token
    if (token >= NTOK) return;

    float acc[NEXP];
    #pragma unroll
    for (int e = 0; e < NEXP; ++e) acc[e] = 0.f;

    for (int j = lane; j < DIM; j += 32) {
        float xr = zr[(size_t)token * DIM + j];
        float xi = zi[(size_t)token * DIM + j];
        #pragma unroll
        for (int e = 0; e < NEXP; ++e) {
            acc[e] += xr * Wg[e * 2 * DIM + j] + xi * Wg[e * 2 * DIM + DIM + j];
        }
    }
    #pragma unroll
    for (int e = 0; e < NEXP; ++e) {
        #pragma unroll
        for (int off = 16; off > 0; off >>= 1)
            acc[e] += __shfl_xor(acc[e], off, 32);
    }
    if (lane == 0) {
        float m = -1e30f;
        #pragma unroll
        for (int e = 0; e < NEXP; ++e) { acc[e] += bg[e]; m = fmaxf(m, acc[e]); }
        float p[NEXP]; float s = 0.f;
        #pragma unroll
        for (int e = 0; e < NEXP; ++e) { p[e] = __expf(acc[e] - m); s += p[e]; }
        float inv = 1.f / s;
        #pragma unroll
        for (int e = 0; e < NEXP; ++e)
            gates[(size_t)token * NEXP + e] = p[e] * inv;
    }
}

// ============ main kernel: gate-fused complex MoE GEMM (bf16 WMMA) ============
// Tile: 64 tokens x 128 outputs per workgroup (8 waves; wave = 16 tok x 64 out).
// Software-pipelined: global loads for stage i+1 issue under the WMMAs of
// stage i; double-buffered LDS -> one barrier per stage.
// For each expert e: a=g*zr, b=g*zi, bn=-b (sign-flip), accumulate
//   out_r += a@Wr^T + bn@Wi^T ; out_i += b@Wr^T + a@Wi^T.
__global__ __launch_bounds__(256) void moe_kernel(
    const float* __restrict__ zr, const float* __restrict__ zi,
    const float* __restrict__ Wr, const float* __restrict__ Wi,
    const float* __restrict__ gates, float* __restrict__ out)
{
    constexpr int BM = 64, BN = 128, LDA = 40;  // 80B row stride: 16B-aligned, conflict-free
    constexpr int NIT = NEXP * (DIM / 32);      // 256 pipeline stages
    __shared__ __align__(16) u16 sA [2][BM * LDA];  //  g * z_real
    __shared__ __align__(16) u16 sB [2][BM * LDA];  //  g * z_imag
    __shared__ __align__(16) u16 sBN[2][BM * LDA];  // -g * z_imag
    __shared__ __align__(16) u16 sWr[2][BN * LDA];
    __shared__ __align__(16) u16 sWi[2][BN * LDA];
    __shared__ __align__(16) float sG[BM][NEXP];    // all gates for this token tile

    const int t    = threadIdx.x;
    const int lane = t & 31;
    const int wave = t >> 5;
    const int tokBase = blockIdx.x * BM;
    const int oBase   = blockIdx.y * BN;

    // wave -> 16x64 subtile
    const int wm = wave & 3;
    const int wn = wave >> 2;
    const int r  = lane & 15;
    const int kbA = (lane < 16) ? 0 : 8;
    const int kbB = (lane < 16) ? 0 : 8;

    // staging work split
    const int arow = t >> 2, ac8  = (t & 3) * 8;   // 64 rows x 32 cols, 8 f32/thread
    const int wrow = t >> 1, wc16 = (t & 1) * 16;  // 128 rows x 32 cols, 16 f32/thread

    // ---- load all gates for this tile once ----
    if (t < 128) {
        float4 gv = *(const float4*)(gates + (size_t)(tokBase + (t >> 1)) * NEXP + (t & 1) * 4);
        *(float4*)&sG[t >> 1][(t & 1) * 4] = gv;
    }

    v8f accR[4], accI[4];
    #pragma unroll
    for (int j = 0; j < 4; ++j) {
        accR[j] = (v8f)(0.0f);
        accI[j] = (v8f)(0.0f);
    }

    // prefetch registers for pipeline stage i+1
    float4 pr0, pr1, pi0, pi1;              // activations
    float4 pw0, pw1, pw2, pw3;              // Wr tile
    float4 pv0, pv1, pv2, pv3;              // Wi tile

    auto load_stage = [&](int it) {
        const int e  = it >> 5;
        const int kc = (it & 31) * 32;
        const float4* pzr = (const float4*)(zr + (size_t)(tokBase + arow) * DIM + kc + ac8);
        const float4* pzi = (const float4*)(zi + (size_t)(tokBase + arow) * DIM + kc + ac8);
        pr0 = pzr[0]; pr1 = pzr[1];
        pi0 = pzi[0]; pi1 = pzi[1];
        const size_t wbase = ((size_t)e * DIM + (oBase + wrow)) * DIM + kc + wc16;
        const float4* pwr = (const float4*)(Wr + wbase);
        const float4* pwi = (const float4*)(Wi + wbase);
        pw0 = pwr[0]; pw1 = pwr[1]; pw2 = pwr[2]; pw3 = pwr[3];
        pv0 = pwi[0]; pv1 = pwi[1]; pv2 = pwi[2]; pv3 = pwi[3];
    };

    auto store_stage = [&](int buf, int e) {
        const float g = sG[arow][e];
        uint4 av = make_uint4(pack_bf2(g * pr0.x, g * pr0.y), pack_bf2(g * pr0.z, g * pr0.w),
                              pack_bf2(g * pr1.x, g * pr1.y), pack_bf2(g * pr1.z, g * pr1.w));
        uint4 bv = make_uint4(pack_bf2(g * pi0.x, g * pi0.y), pack_bf2(g * pi0.z, g * pi0.w),
                              pack_bf2(g * pi1.x, g * pi1.y), pack_bf2(g * pi1.z, g * pi1.w));
        // bf16 negation == sign-bit flip of both packed halves
        uint4 nv = make_uint4(bv.x ^ 0x80008000u, bv.y ^ 0x80008000u,
                              bv.z ^ 0x80008000u, bv.w ^ 0x80008000u);
        *(uint4*)&sA [buf][arow * LDA + ac8] = av;
        *(uint4*)&sB [buf][arow * LDA + ac8] = bv;
        *(uint4*)&sBN[buf][arow * LDA + ac8] = nv;
        uint4* dr = (uint4*)&sWr[buf][wrow * LDA + wc16];
        uint4* di = (uint4*)&sWi[buf][wrow * LDA + wc16];
        dr[0] = make_uint4(pack_bf2(pw0.x, pw0.y), pack_bf2(pw0.z, pw0.w),
                           pack_bf2(pw1.x, pw1.y), pack_bf2(pw1.z, pw1.w));
        dr[1] = make_uint4(pack_bf2(pw2.x, pw2.y), pack_bf2(pw2.z, pw2.w),
                           pack_bf2(pw3.x, pw3.y), pack_bf2(pw3.z, pw3.w));
        di[0] = make_uint4(pack_bf2(pv0.x, pv0.y), pack_bf2(pv0.z, pv0.w),
                           pack_bf2(pv1.x, pv1.y), pack_bf2(pv1.z, pv1.w));
        di[1] = make_uint4(pack_bf2(pv2.x, pv2.y), pack_bf2(pv2.z, pv2.w),
                           pack_bf2(pv3.x, pv3.y), pack_bf2(pv3.z, pv3.w));
    };

    load_stage(0);

    int buf = 0;
    for (int it = 0; it < NIT; ++it) {
        store_stage(buf, it >> 5);
        __syncthreads();   // does not cover the sG fill: store_stage read sG above

        if (it + 1 < NIT) load_stage(it + 1);   // global loads overlap the WMMAs below

        // ---- WMMA compute for stage it ----
        v16bf fa  = ldfragA(&sA [buf][(wm * 16 + r) * LDA + kbA]);
        v16bf fb  = ldfragA(&sB [buf][(wm * 16 + r) * LDA + kbA]);
        v16bf fbn = ldfragA(&sBN[buf][(wm * 16 + r) * LDA + kbA]);
        #pragma unroll
        for (int j = 0; j < 4; ++j) {
            const int wofs = (wn * 64 + j * 16 + r) * LDA + kbB;
            v16bf fwr = ldfragB(&sWr[buf][wofs]);
            v16bf fwi = ldfragB(&sWi[buf][wofs]);
            accR[j] = __builtin_amdgcn_wmma_f32_16x16x32_bf16(false, fa,  false, fwr, (short)0, accR[j], false, false);
            accR[j] = __builtin_amdgcn_wmma_f32_16x16x32_bf16(false, fbn, false, fwi, (short)0, accR[j], false, false);
            accI[j] = __builtin_amdgcn_wmma_f32_16x16x32_bf16(false, fb,  false, fwr, (short)0, accI[j], false, false);
            accI[j] = __builtin_amdgcn_wmma_f32_16x16x32_bf16(false, fa,  false, fwi, (short)0, accI[j], false, false);
        }
        buf ^= 1;
        // no trailing barrier: next stage stores to the other LDS buffer; the
        // barrier at the top of iteration it+1 orders compute(it) vs store(it+2).
    }

    // ---- write out: C/D layout: VGPR v -> M = v + (lane<16?0:8), N = lane&15 ----
    float* outR = out;
    float* outI = out + (size_t)NTOK * DIM;
    const int mhalf = (lane < 16) ? 0 : 8;
    #pragma unroll
    for (int j = 0; j < 4; ++j) {
        const int n = oBase + wn * 64 + j * 16 + r;
        #pragma unroll
        for (int v = 0; v < 8; ++v) {
            const size_t row = (size_t)(tokBase + wm * 16 + mhalf + v);
            outR[row * DIM + n] = accR[j][v];
            outI[row * DIM + n] = accI[j][v];
        }
    }
}

extern "C" void kernel_launch(void* const* d_in, const int* in_sizes, int n_in,
                              void* d_out, int out_size, void* d_ws, size_t ws_size,
                              hipStream_t stream) {
    const float* zr = (const float*)d_in[0];
    const float* zi = (const float*)d_in[1];
    const float* Wg = (const float*)d_in[2];
    const float* bg = (const float*)d_in[3];
    const float* Wr = (const float*)d_in[4];
    const float* Wi = (const float*)d_in[5];
    float* out   = (float*)d_out;
    float* gates = (float*)d_ws;   // [NTOK, NEXP] fp32 = 256 KB

    gate_kernel<<<NTOK / 8, 256, 0, stream>>>(zr, zi, Wg, bg, gates);

    dim3 grid(NTOK / 64, DIM / 128);
    moe_kernel<<<grid, 256, 0, stream>>>(zr, zi, Wr, Wi, gates, out);
}
